// ArcFaceLoss_90520730730788
// MI455X (gfx1250) — matrix operand order repaired and checked by reading, not compiled, not run on hardware
//
#include <hip/hip_runtime.h>
#include <math.h>

// ---------------- problem constants ----------------
#define K_DIM   512
#define N_CLS   100000
#define BATCH   2048
#define NT      128                     // classes per workgroup
#define LDB     520                     // padded LDS k-stride (elements) -> conflict-free banks
#define NSUB    (N_CLS / 16)            // 6250 16-class sub-tiles
#define NBLK    ((NSUB + 7) / 8)        // 782 workgroups / class blocks
#define S_SCALE 64.0f
#define COS_M_C 0.8775825618903728f
#define SIN_M_C 0.479425538604203f
#define TH_C    (-0.8775825618903728f)
#define MM_C    0.23971276930210156f

typedef __attribute__((ext_vector_type(16))) __bf16 bf16x16;
typedef __attribute__((ext_vector_type(8)))  __bf16 bf16x8;
typedef __attribute__((ext_vector_type(4)))  __bf16 bf16x4;
typedef __attribute__((ext_vector_type(8)))  float  floatx8;

static __device__ __forceinline__ bf16x8 lds_ld8(const unsigned short* p) {
    return __builtin_bit_cast(bf16x8, *(const uint4*)p);
}
static __device__ __forceinline__ bf16x16 cat8(bf16x8 lo, bf16x8 hi) {
    return __builtin_shufflevector(lo, hi, 0,1,2,3,4,5,6,7,8,9,10,11,12,13,14,15);
}
static __device__ __forceinline__ bf16x4 cvt4(float4 v) {
    bf16x4 h;
    h[0] = (__bf16)v.x; h[1] = (__bf16)v.y; h[2] = (__bf16)v.z; h[3] = (__bf16)v.w;
    return h;
}

// ---------------- kernel 0: rnorm[c] = 1/max(||w_c||, 1e-12) ----------------
__global__ __launch_bounds__(256) void rnorm_kernel(const float* __restrict__ w,
                                                    float* __restrict__ rnorm) {
    int row  = blockIdx.x * 8 + (threadIdx.x >> 5);   // one wave per class row
    int lane = threadIdx.x & 31;
    const float4* r = (const float4*)(w + (size_t)row * K_DIM);
    float sum = 0.f;
#pragma unroll
    for (int i = 0; i < 4; ++i) {
        float4 v = r[lane + i * 32];
        sum += v.x*v.x + v.y*v.y + v.z*v.z + v.w*v.w;
    }
#pragma unroll
    for (int m = 1; m <= 16; m <<= 1) sum += __shfl_xor(sum, m, 32);
    if (lane == 0) rnorm[row] = 1.0f / fmaxf(sqrtf(sum), 1e-12f);
}

// ---------------- kernel 1: embeddings fp32 -> bf16 (L2-resident A) ----------------
__global__ __launch_bounds__(256) void emb_to_bf16(const float4* __restrict__ in,
                                                   bf16x4* __restrict__ out) {
    int i = blockIdx.x * blockDim.x + threadIdx.x;    // 262144 float4 groups
    out[i] = cvt4(in[i]);
}

// ---------------- kernel 2: exact fp32 target logit S*phi(cos(b, label_b)) ----------------
__global__ __launch_bounds__(256) void tgt_kernel(const float* __restrict__ emb,
                                                  const float* __restrict__ weight,
                                                  const float* __restrict__ rnorm,
                                                  const int* __restrict__ labels,
                                                  float* __restrict__ tgt) {
    int b    = blockIdx.x * 8 + (threadIdx.x >> 5);   // one wave per batch row
    int lane = threadIdx.x & 31;
    int lbl  = labels[b];
    const float4* e = (const float4*)(emb    + (size_t)b   * K_DIM);
    const float4* w = (const float4*)(weight + (size_t)lbl * K_DIM);
    float sum = 0.f;
#pragma unroll
    for (int i = 0; i < 4; ++i) {
        float4 a = e[lane + i * 32];
        float4 c = w[lane + i * 32];
        sum += a.x*c.x + a.y*c.y + a.z*c.z + a.w*c.w;
    }
#pragma unroll
    for (int m = 1; m <= 16; m <<= 1) sum += __shfl_xor(sum, m, 32);
    if (lane == 0) {
        float cosv = sum * rnorm[lbl];
        float s2   = fminf(fmaxf(1.0f - cosv * cosv, 1e-7f), 1.0f);
        float phi  = cosv * COS_M_C - sqrtf(s2) * SIN_M_C;
        phi = (cosv > TH_C) ? phi : (cosv - MM_C);
        tgt[b] = S_SCALE * phi;
    }
}

// ---------------- kernel 3: fused WMMA GEMM + ArcFace + partial sum-exp ----------------
// grid = NBLK workgroups; each owns 128 classes (B resident in LDS, weight read from
// HBM exactly once), loops over all 128 batch tiles of 16 rows.
// Logits are bounded by ~|S|=64, so exp() cannot overflow fp32: no max-tracking needed.
__global__ __launch_bounds__(256) void arc_main(const uint4*  __restrict__ embq,     // bf16 embeddings, 8 per uint4
                                                const float*  __restrict__ weight,
                                                const float*  __restrict__ rnorm,
                                                const int*    __restrict__ labels,
                                                float*        __restrict__ partials) {
    extern __shared__ char smem[];
    unsigned short* ldsB = (unsigned short*)smem;                       // [NT][LDB] bf16
    unsigned short* ldsA = (unsigned short*)(smem + NT * LDB * 2);      // [16][LDB] bf16
    float*          sred = (float*)(smem + (NT + 16) * LDB * 2);        // [8 waves][16 rows]

    const int tid   = threadIdx.x;
    const int g     = blockIdx.x;
    const int w     = tid >> 5;
    const int lane  = tid & 31;
    const int laneN = lane & 15;
    const int half  = lane >> 4;          // 0: lanes 0-15, 1: lanes 16-31

    // ---- stage B block: 128 classes x 512 K, fp32 -> bf16, into LDS (coalesced) ----
    const int c0g = g * NT;
#pragma unroll 4
    for (int i = 0; i < 64; ++i) {        // 64 * 256 thr * 4 floats = 65536 elements
        int idx = i * 1024 + tid * 4;
        int c = idx >> 9, k = idx & 511;
        float4 v = make_float4(0.f, 0.f, 0.f, 0.f);
        int cls = c0g + c;
        if (cls < N_CLS) v = *(const float4*)(weight + (size_t)cls * K_DIM + k);
        *(bf16x4*)(&ldsB[c * LDB + k]) = cvt4(v);
    }
    __syncthreads();

    const int  s_tile   = g * 8 + w;            // this wave's 16-class sub-tile
    const bool valid    = (s_tile < NSUB);      // N_CLS % 16 == 0, so whole tile valid
    const int  cls_lane = s_tile * 16 + laneN;  // class owned by this lane's N column
    const float rn      = valid ? rnorm[cls_lane] : 0.0f;

    // A-loader mapping for this thread (32 contiguous bf16 = 4 x 16B)
    const int a_idx = tid * 32;
    const int a_row = a_idx >> 9, a_k = a_idx & 511;

    for (int mt = 0; mt < BATCH / 16; ++mt) {
        // ---- cooperative A-tile load: 16 rows x 512 bf16 from global bf16 embeddings ----
        {
            const uint4* src = embq + (((size_t)(mt * 16 + a_row) * K_DIM + a_k) >> 3);
#pragma unroll
            for (int j = 0; j < 4; ++j)
                *(uint4*)(&ldsA[a_row * LDB + a_k + j * 8]) = src[j];
        }
        // prefetch next iteration's A-tile slice into near caches (global_prefetch_b8)
        if (mt + 1 < BATCH / 16) {
            const char* np = (const char*)embq +
                             (((size_t)((mt + 1) * 16 + a_row) * K_DIM + a_k) << 1);
            __builtin_prefetch(np, 0, 1);
        }
        __syncthreads();

        // ---- 16x16 tile GEMM over K=512: 16 x v_wmma_f32_16x16x32_bf16 ----
        floatx8 acc = {0.f, 0.f, 0.f, 0.f, 0.f, 0.f, 0.f, 0.f};
        const unsigned short* aRow = &ldsA[laneN * LDB];
        const unsigned short* bRow = &ldsB[(w * 16 + laneN) * LDB];
#pragma unroll
        for (int t = 0; t < 16; ++t) {
            int ka = t * 32 + half * 8;      // A: lane holds K [ka..ka+7] and [ka+16..ka+23]
            int kb = t * 32 + half * 8;      // B: lane holds K [kb..kb+7] and [kb+16..kb+23]
            bf16x16 Af = cat8(lds_ld8(aRow + ka), lds_ld8(aRow + ka + 16));
            bf16x16 Bf = cat8(lds_ld8(bRow + kb), lds_ld8(bRow + kb + 16));
            acc = __builtin_amdgcn_wmma_f32_16x16x32_bf16(false, Af, false, Bf,
                                                          (short)0, acc, false, false);
        }

        // ---- epilogue: rnorm scale, ArcFace phi on label column, row-wise sum(exp) ----
#pragma unroll
        for (int r = 0; r < 8; ++r) {
            int row = mt * 16 + r + half * 8;      // C layout: VGPR r -> M = r + 8*half
            float e;
            if (valid) {
                float cosv = acc[r] * rn;
                float logit;
                int lbl = labels[row];
                if (lbl == cls_lane) {
                    float s2  = fminf(fmaxf(1.0f - cosv * cosv, 1e-7f), 1.0f);
                    float phi = cosv * COS_M_C - sqrtf(s2) * SIN_M_C;
                    phi = (cosv > TH_C) ? phi : (cosv - MM_C);
                    logit = S_SCALE * phi;
                } else {
                    logit = S_SCALE * cosv;
                }
                e = expf(logit);                   // |logit| <~ 64 -> always finite in fp32
            } else {
                e = 0.0f;
            }
            // sum across the 16 lanes of this half-wave (the 16 classes of row `row`)
#pragma unroll
            for (int msk = 1; msk <= 8; msk <<= 1) e += __shfl_xor(e, msk, 32);
            if (laneN == 0) sred[w * 16 + r + half * 8] = e;
        }
        __syncthreads();

        // ---- combine the 8 waves' partial sums for these 16 rows ----
        if (tid < 16) {
            float S = 0.0f;
#pragma unroll
            for (int w2 = 0; w2 < 8; ++w2) S += sred[w2 * 16 + tid];
            partials[(size_t)(mt * 16 + tid) * NBLK + g] = S;
        }
        __syncthreads();   // protect sred and ldsA for next iteration
    }
}

// ---------------- kernel 4: per-row sum of 782 partials -> log(sum) - tgt ----------------
__global__ __launch_bounds__(256) void row_lse_kernel(const float* __restrict__ partials,
                                                      const float* __restrict__ tgt,
                                                      float*       __restrict__ row_loss) {
    int b    = blockIdx.x * 8 + (threadIdx.x >> 5);   // one wave per batch row
    int lane = threadIdx.x & 31;
    float s = 0.0f;
    for (int j = lane; j < NBLK; j += 32) s += partials[(size_t)b * NBLK + j];
#pragma unroll
    for (int off = 16; off >= 1; off >>= 1) s += __shfl_xor(s, off, 32);
    if (lane == 0) row_loss[b] = logf(s) - tgt[b];
}

// ---------------- kernel 5: deterministic mean over 2048 rows ----------------
__global__ __launch_bounds__(256) void final_mean(const float* __restrict__ row_loss,
                                                  float* __restrict__ out) {
    __shared__ float red[256];
    float s = 0.0f;
#pragma unroll
    for (int i = 0; i < 8; ++i) s += row_loss[threadIdx.x + i * 256];
    red[threadIdx.x] = s;
    __syncthreads();
    for (int h = 128; h > 0; h >>= 1) {
        if ((int)threadIdx.x < h) red[threadIdx.x] += red[threadIdx.x + h];
        __syncthreads();
    }
    if (threadIdx.x == 0) out[0] = red[0] * (1.0f / (float)BATCH);
}

// ---------------- launch ----------------
extern "C" void kernel_launch(void* const* d_in, const int* in_sizes, int n_in,
                              void* d_out, int out_size, void* d_ws, size_t ws_size,
                              hipStream_t stream) {
    const float* emb    = (const float*)d_in[0];   // [2048, 512] fp32
    const int*   labels = (const int*)d_in[1];     // [2048] int
    const float* weight = (const float*)d_in[2];   // [100000, 512] fp32
    float*       out    = (float*)d_out;           // scalar loss

    // workspace layout (all 16B aligned)
    char* ws = (char*)d_ws;
    float*          rnorm    = (float*)(ws);                              //   400,000 B
    float*          tgt      = (float*)(ws + 400000);                     //     8,192 B
    unsigned short* embbf    = (unsigned short*)(ws + 408192);            // 2,097,152 B
    float*          partials = (float*)(ws + 2505344);                    // 6,406,144 B
    float*          row_loss = (float*)(ws + 8911488);                    //     8,192 B

    rnorm_kernel<<<N_CLS / 8, 256, 0, stream>>>(weight, rnorm);
    emb_to_bf16<<<(BATCH * K_DIM / 4) / 256, 256, 0, stream>>>((const float4*)emb, (bf16x4*)embbf);
    tgt_kernel<<<BATCH / 8, 256, 0, stream>>>(emb, weight, rnorm, labels, tgt);

    size_t smem = (size_t)(NT + 16) * LDB * 2 + 8 * 16 * sizeof(float);   // 150,272 B
    arc_main<<<NBLK, 256, smem, stream>>>((const uint4*)embbf, weight, rnorm, labels, partials);

    row_lse_kernel<<<BATCH / 8, 256, 0, stream>>>(partials, tgt, row_loss);
    final_mean<<<1, 256, 0, stream>>>(row_loss, out);
}
